// ResidualHyperResNet_86758339379730
// MI455X (gfx1250) — matrix operand order repaired and checked by reading, not compiled, not run on hardware
//
#include <hip/hip_runtime.h>

typedef _Float16 half_t;
typedef __attribute__((ext_vector_type(16))) _Float16 v16h;
typedef __attribute__((ext_vector_type(8)))  _Float16 v8h;
typedef __attribute__((ext_vector_type(8)))  float    v8f;

constexpr int BATCH = 512;

__device__ __forceinline__ v16h ld_pair(const half_t* p, int off2){
  v8h lo = *(const v8h*)p;
  v8h hi = *(const v8h*)(p + off2);
  return __builtin_shufflevector(lo, hi, 0,1,2,3,4,5,6,7,8,9,10,11,12,13,14,15);
}

// ============================ small kernels ============================

// static conv weights: [outc][INC][3][3] f32 -> [outc][9*ICP] f16, f = tap*ICP + ic (zero-padded ic)
template<int INC, int ICP>
__global__ void pack_perm_kernel(const float* __restrict__ s, half_t* __restrict__ d, int outc){
  int i = blockIdx.x * 256 + threadIdx.x;
  int total = outc * 9 * ICP;
  if (i >= total) return;
  int oc = i / (9 * ICP); int r = i % (9 * ICP);
  int tap = r / ICP, ic = r % ICP;
  float v = (ic < INC) ? s[(oc * INC + ic) * 9 + tap] : 0.0f;
  d[i] = (half_t)v;
}

__global__ __launch_bounds__(128) void trunk_kernel(
    const float* __restrict__ z, const float* __restrict__ W1, const float* __restrict__ b1,
    const float* __restrict__ W2, const float* __restrict__ b2, half_t* __restrict__ h16){
  __shared__ float sZ[100];
  __shared__ float sH[128];
  int b = blockIdx.x, t = threadIdx.x;
  if (t < 100) sZ[t] = z[b * 100 + t];
  __syncthreads();
  float a = b1[t];
  for (int k = 0; k < 100; ++k) a = fmaf(sZ[k], W1[t * 100 + k], a);
  sH[t] = fmaxf(a, 0.0f);
  __syncthreads();
  float c = b2[t];
  for (int k = 0; k < 128; ++k) c = fmaf(sH[k], W2[t * 128 + k], c);
  h16[(size_t)b * 128 + t] = (half_t)fmaxf(c, 0.0f);
}

__global__ void avgpool_kernel(const float* __restrict__ in, float* __restrict__ feat){
  int b = blockIdx.x, c = threadIdx.x;               // 64 threads
  const float* p = in + ((size_t)b * 64 + c) * 64;   // [b][64][8*8]
  float s = 0.f;
  for (int i = 0; i < 64; ++i) s += p[i];
  feat[b * 64 + c] = s * (1.0f / 64.0f);
}

__global__ __launch_bounds__(128) void fc_kernel(const half_t* __restrict__ wt,
                                                 const float* __restrict__ feat,
                                                 float* __restrict__ out){
  __shared__ float sF[64];
  int b = blockIdx.x, t = threadIdx.x;
  if (t < 64) sF[t] = feat[b * 64 + t];
  __syncthreads();
  if (t < 100){
    const half_t* w = wt + ((size_t)b * 100 + t) * 64;
    float s = 0.f;
    for (int i = 0; i < 64; ++i) s = fmaf((float)w[i], sF[i], s);
    out[b * 100 + t] = s;
  }
}

// ===================== hyper weight generation (WMMA) =====================
// dw = h · headT (K=128 via v_wmma_f32_16x16x32_f16), standardized over fan_in in LDS,
// written f16 with K re-ordered f_dst = tap*INC + ic (KSQ==1 -> identity).
template<int FANIN, int OUTC, int KSQ>
__global__ __launch_bounds__(128) void wgen_kernel(
    const half_t* __restrict__ h16,   // [B][128]
    const float*  __restrict__ head,  // [OUTC*FANIN][128]
    const float*  __restrict__ base,  // [OUTC*FANIN]
    half_t* __restrict__ wt)          // [B][OUTC][FANIN]  (f16, permuted K)
{
  static_assert(FANIN % 16 == 0, "fan_in multiple of 16");
  constexpr int NT = FANIN / 16;
  constexpr int INCQ = FANIN / KSQ;
  __shared__ alignas(16) half_t sH[16 * 128];
  __shared__ float  sDw[16 * FANIN];
  __shared__ float  sSum[16], sSqs[16];

  int oc = blockIdx.x % OUTC;
  int b0 = (blockIdx.x / OUTC) * 16;
  int tid = threadIdx.x, lane = tid & 31, wv = tid >> 5;

  for (int i8 = tid; i8 < (16 * 128) / 8; i8 += 128)
    *(v8h*)&sH[i8 * 8] = *(const v8h*)&h16[(size_t)b0 * 128 + i8 * 8];
  __syncthreads();

  const float* hd = head + (size_t)oc * FANIN * 128;
  int n   = lane & 15;
  int kbB = (lane >> 4) << 4;
  int kbA = (lane >> 4) << 4;
  int m   = lane & 15;

  for (int nt = wv; nt < NT; nt += 4){
    v8f C = {};
    const float* hrow = hd + (size_t)(nt * 16 + n) * 128;
    for (int kc = 0; kc < 4; ++kc){
      v16h A = ld_pair(&sH[m * 128 + kc * 32 + kbA], 8);
      const float4* h4 = (const float4*)(hrow + kc * 32 + kbB);
      float4 q0 = h4[0], q1 = h4[1], q2 = h4[2], q3 = h4[3];
      v16h Bv;
      Bv[0]=(half_t)q0.x; Bv[1]=(half_t)q0.y; Bv[2]=(half_t)q0.z; Bv[3]=(half_t)q0.w;
      Bv[4]=(half_t)q1.x; Bv[5]=(half_t)q1.y; Bv[6]=(half_t)q1.z; Bv[7]=(half_t)q1.w;
      Bv[8]=(half_t)q2.x; Bv[9]=(half_t)q2.y; Bv[10]=(half_t)q2.z; Bv[11]=(half_t)q2.w;
      Bv[12]=(half_t)q3.x; Bv[13]=(half_t)q3.y; Bv[14]=(half_t)q3.z; Bv[15]=(half_t)q3.w;
      C = __builtin_amdgcn_wmma_f32_16x16x32_f16(false, A, false, Bv, (short)0, C, false, false);
    }
    #pragma unroll
    for (int r = 0; r < 8; ++r){
      int s = r + ((lane >> 4) << 3);     // sample row (M)
      sDw[s * FANIN + nt * 16 + n] = C[r];
    }
  }
  __syncthreads();

  { // per-sample mean/var over FANIN: 16 samples x 8 threads
    int s = tid >> 3, st = tid & 7;
    float sum = 0.f, sq = 0.f;
    for (int f = st; f < FANIN; f += 8){ float v = sDw[s * FANIN + f]; sum += v; sq += v * v; }
    for (int off = 4; off; off >>= 1){ sum += __shfl_xor(sum, off, 8); sq += __shfl_xor(sq, off, 8); }
    if (st == 0){ sSum[s] = sum; sSqs[s] = sq; }
  }
  __syncthreads();

  const float inv   = 1.0f / (float)FANIN;
  const float scale = sqrtf(2.0f / (float)FANIN);
  const float is2   = 0.70710678118654752f;
  for (int i = tid; i < 16 * FANIN; i += 128){
    int s = i / FANIN, fd = i % FANIN;           // fd: destination (permuted) index
    int tap = fd / INCQ, ic = fd % INCQ;
    int fs = ic * KSQ + tap;                     // source (reference) index
    float mu = sSum[s] * inv;
    float var = sSqs[s] * inv - mu * mu;
    float rs = rsqrtf(var + 1e-5f);
    float v = (sDw[s * FANIN + fs] - mu) * rs * scale;
    float w = (base[(size_t)oc * FANIN + fs] + v) * is2;
    wt[((size_t)(b0 + s) * OUTC + oc) * FANIN + fd] = (half_t)w;
  }
}

// ===================== implicit-GEMM conv (WMMA) + fused GN =====================
// Channel-last f16 activations in LDS; K flattened f = tap*ICP + ic so every WMMA
// operand is two contiguous ds_load_b128. Zero-padded K chunks select a zero slot.
template<int INC, int ICP, int OUTC, int HIN, int WIN, int STR, int KS,
         bool GN, bool RES, bool RELU, bool PS>
__global__ __launch_bounds__(256) void hconv_kernel(
    const float*  __restrict__ in,    // [B][INC][HIN*WIN]
    const half_t* __restrict__ wt,    // [B?][OUTC][FANF] (permuted K)
    const float*  __restrict__ gamma, const float* __restrict__ beta,
    const float*  __restrict__ res,   // [B][OUTC][NPIX] or null
    float* __restrict__ out)          // [B][OUTC][NPIX]
{
  constexpr int PAD = KS / 2;
  constexpr int HP = HIN + 2 * PAD, WP = WIN + 2 * PAD;
  constexpr int HOUT = HIN / STR, WOUT = WIN / STR;
  constexpr int NPIX = HOUT * WOUT;
  constexpr int FANF = ICP * KS * KS;
  constexpr int FANP = ((FANF + 31) / 32) * 32;
  constexpr int NKC = FANP / 32;
  constexpr int NOCT = OUTC / 16, NPT = NPIX / 16, NTILES = NOCT * NPT;
  constexpr int NW = 8;
  constexpr int CPG = OUTC / 8;
  constexpr int ZOFF = HP * WP * ICP;          // 16-half zero slot
  static_assert(ICP % 16 == 0 && OUTC % 16 == 0 && NPIX % 16 == 0, "tile dims");

  __shared__ alignas(16) half_t sAct[HP * WP * ICP + 16];
  __shared__ alignas(16) half_t sWt[OUTC * FANP];
  __shared__ float  sOut[GN ? OUTC * NPIX : 1];
  __shared__ float  sSum[8], sSqs[8];

  int b = blockIdx.x;
  int tid = threadIdx.x, lane = tid & 31, wv = tid >> 5;

  { v8h z = {};
    for (int i8 = tid; i8 < (HP * WP * ICP + 16) / 8; i8 += 256) *(v8h*)&sAct[i8 * 8] = z; }
  __syncthreads();

  // activations: global [ic][y][x] f32 -> LDS [(y+PAD)*WP + x+PAD][ic] f16 (b128 stores)
  const float* inp = in + (size_t)b * INC * HIN * WIN;
  for (int p0 = tid; p0 < HIN * WIN; p0 += 256){
    int y = p0 / WIN, x = p0 % WIN;
    int pb = ((y + PAD) * WP + (x + PAD)) * ICP;
    #pragma unroll
    for (int c8 = 0; c8 < ICP / 8; ++c8){
      v8h v;
      #pragma unroll
      for (int j = 0; j < 8; ++j){
        int ic = c8 * 8 + j;
        v[j] = (ic < INC) ? (half_t)inp[ic * HIN * WIN + p0] : (half_t)0;
      }
      *(v8h*)&sAct[pb + c8 * 8] = v;
    }
  }
  // weights: [oc][FANF] f16 -> LDS zero-padded to FANP (b128 copies)
  const half_t* wp = wt + (PS ? (size_t)b * OUTC * FANF : 0);
  for (int i8 = tid; i8 < (OUTC * FANP) / 8; i8 += 256){
    int oc = (i8 * 8) / FANP, f = (i8 * 8) % FANP;
    v8h v = {};
    if (f < FANF) v = *(const v8h*)(wp + (size_t)oc * FANF + f);
    *(v8h*)&sWt[i8 * 8] = v;
  }
  __syncthreads();

  int n = lane & 15;
  int kbB = (lane >> 4) << 4;
  int kbA = (lane >> 4) << 4;
  float* op = out + (size_t)b * OUTC * NPIX;

  for (int t = wv; t < NTILES; t += NW){
    int oct = t / NPT, pt = t % NPT;
    int p = pt * 16 + n;
    int py = p / WOUT, px = p % WOUT;
    int ocA = oct * 16 + (lane & 15);
    const half_t* arow = &sWt[ocA * FANP];
    v8f C = {};
    for (int kc = 0; kc < NKC; ++kc){
      v16h A = ld_pair(arow + kc * 32 + kbA, 8);
      int fg = kc * 32 + kbB;
      int tap = fg / ICP, ic0 = fg % ICP;
      int ky = tap / KS, kx = tap % KS;
      int baddr = ((py * STR + ky) * WP + (px * STR + kx)) * ICP + ic0;
      int bsel = (fg < FANF) ? baddr : ZOFF;
      v16h Bv = ld_pair(&sAct[bsel], 8);
      C = __builtin_amdgcn_wmma_f32_16x16x32_f16(false, A, false, Bv, (short)0, C, false, false);
    }
    if constexpr (GN){
      #pragma unroll
      for (int r = 0; r < 8; ++r){
        int oc = oct * 16 + r + ((lane >> 4) << 3);
        sOut[oc * NPIX + p] = C[r];
      }
    } else {
      #pragma unroll
      for (int r = 0; r < 8; ++r){
        int oc = oct * 16 + r + ((lane >> 4) << 3);
        op[oc * NPIX + p] = C[r];
      }
    }
  }

  if constexpr (GN){
    __syncthreads();
    { // group wv of 8: reduce CPG*NPIX values with wave32 shuffle
      float s = 0.f, q = 0.f;
      for (int i = lane; i < CPG * NPIX; i += 32){
        float v = sOut[wv * CPG * NPIX + i];
        s += v; q += v * v;
      }
      for (int off = 16; off; off >>= 1){ s += __shfl_xor(s, off, 32); q += __shfl_xor(q, off, 32); }
      if (lane == 0){ sSum[wv] = s; sSqs[wv] = q; }
    }
    __syncthreads();
    const float cnt = 1.0f / (float)(CPG * NPIX);
    const float* rp = RES ? (res + (size_t)b * OUTC * NPIX) : nullptr;
    for (int i = tid; i < OUTC * NPIX; i += 256){
      int oc = i / NPIX;
      int g = oc / CPG;
      float mu = sSum[g] * cnt;
      float var = sSqs[g] * cnt - mu * mu;
      float rs = rsqrtf(var + 1e-5f);
      float v = (sOut[i] - mu) * rs * gamma[oc] + beta[oc];
      if constexpr (RES) v += rp[i];
      if constexpr (RELU) v = fmaxf(v, 0.0f);
      op[i] = v;
    }
  }
}

// ============================ host launcher ============================

struct HyperP { const float *base1,*head1,*g1,*b1,*base2,*head2,*g2,*b2,*scb,*sch; };
struct Params {
  const float *x,*z,*tW1,*tb1,*tW2,*tb2,*c1w,*n1g,*n1b,*fcb,*fch;
  const float *l1w1[3],*l1g1[3],*l1b1[3],*l1w2[3],*l1g2[3],*l1b2[3];
  HyperP l2[3], l3[3];
};

extern "C" void kernel_launch(void* const* d_in, const int* in_sizes, int n_in,
                              void* d_out, int out_size, void* d_ws, size_t ws_size,
                              hipStream_t stream) {
  auto F = [&](int i){ return (const float*)d_in[i]; };
  Params P{};
  bool orderA = (n_in > 0 && in_sizes[0] == 512 * 3 * 32 * 32);
  if (orderA){ // setup_inputs() insertion order: x, z, params{trunk, conv1_w, ...}
    P.x = F(0); P.z = F(1);
    P.tW1 = F(2); P.tb1 = F(3); P.tW2 = F(4); P.tb2 = F(5);
    P.c1w = F(6); P.n1g = F(7); P.n1b = F(8);
    for (int bk = 0; bk < 3; ++bk){
      int i0 = 9 + bk * 6;
      P.l1w1[bk]=F(i0); P.l1g1[bk]=F(i0+1); P.l1b1[bk]=F(i0+2);
      P.l1w2[bk]=F(i0+3); P.l1g2[bk]=F(i0+4); P.l1b2[bk]=F(i0+5);
    }
    int idx = 27;
    for (int bk = 0; bk < 3; ++bk){
      HyperP& h = P.l2[bk];
      h.base1=F(idx++); h.head1=F(idx++); h.g1=F(idx++); h.b1=F(idx++);
      h.base2=F(idx++); h.head2=F(idx++); h.g2=F(idx++); h.b2=F(idx++);
      if (bk == 0){ h.scb=F(idx++); h.sch=F(idx++); } else { h.scb=nullptr; h.sch=nullptr; }
    }
    for (int bk = 0; bk < 3; ++bk){
      HyperP& h = P.l3[bk];
      h.base1=F(idx++); h.head1=F(idx++); h.g1=F(idx++); h.b1=F(idx++);
      h.base2=F(idx++); h.head2=F(idx++); h.g2=F(idx++); h.b2=F(idx++);
      if (bk == 0){ h.scb=F(idx++); h.sch=F(idx++); } else { h.scb=nullptr; h.sch=nullptr; }
    }
    P.fcb = F(79); P.fch = F(80);
  } else {     // jax sorted-pytree order: params(sorted keys), x, z
    P.c1w = F(0); P.fcb = F(1); P.fch = F(2);
    for (int bk = 0; bk < 3; ++bk){
      int i0 = 3 + bk * 6; // b1,b2,g1,g2,w1,w2
      P.l1b1[bk]=F(i0); P.l1b2[bk]=F(i0+1); P.l1g1[bk]=F(i0+2);
      P.l1g2[bk]=F(i0+3); P.l1w1[bk]=F(i0+4); P.l1w2[bk]=F(i0+5);
    }
    int idx = 21;
    for (int bk = 0; bk < 3; ++bk){ // b1,b2,base1,base2,g1,g2,head1,head2[,sc_base,sc_head]
      HyperP& h = P.l2[bk];
      h.b1=F(idx++); h.b2=F(idx++); h.base1=F(idx++); h.base2=F(idx++);
      h.g1=F(idx++); h.g2=F(idx++); h.head1=F(idx++); h.head2=F(idx++);
      if (bk == 0){ h.scb=F(idx++); h.sch=F(idx++); } else { h.scb=nullptr; h.sch=nullptr; }
    }
    for (int bk = 0; bk < 3; ++bk){
      HyperP& h = P.l3[bk];
      h.b1=F(idx++); h.b2=F(idx++); h.base1=F(idx++); h.base2=F(idx++);
      h.g1=F(idx++); h.g2=F(idx++); h.head1=F(idx++); h.head2=F(idx++);
      if (bk == 0){ h.scb=F(idx++); h.sch=F(idx++); } else { h.scb=nullptr; h.sch=nullptr; }
    }
    P.n1b = F(73); P.n1g = F(74);
    P.tW1 = F(75); P.tb1 = F(76); P.tW2 = F(77); P.tb2 = F(78);
    P.x = F(79); P.z = F(80);
  }

  // ---- workspace carve-up ----
  char* base = (char*)d_ws;
  size_t off = 0;
  auto carve = [&](size_t bytes)->char*{
    char* p = base + off;
    off = (off + bytes + 255) & ~(size_t)255;
    return p;
  };
  half_t* h16  = (half_t*)carve((size_t)BATCH * 128 * 2);
  float*  bufA = (float*) carve((size_t)BATCH * 16 * 1024 * 4);
  float*  bufB = (float*) carve((size_t)BATCH * 16 * 1024 * 4);
  float*  bufC = (float*) carve((size_t)BATCH * 32 * 256 * 4);
  half_t* wt   = (half_t*)carve((size_t)BATCH * 36864 * 2);
  half_t* wtfc = (half_t*)carve((size_t)BATCH * 100 * 64 * 2);
  float*  feat = (float*) carve((size_t)BATCH * 64 * 4);
  half_t* pkC1 = (half_t*)carve((size_t)16 * 144 * 2);       // conv1, ICP=16
  half_t* pkL1 = (half_t*)carve((size_t)6 * 16 * 144 * 2);   // layer1 convs, ICP=16
  if (off > ws_size) return; // insufficient scratch; deterministic no-op

  // ---- pack + permute static conv weights to f16 channel-last K order ----
  pack_perm_kernel<3,16><<<9, 256, 0, stream>>>(P.c1w, pkC1, 16);
  for (int bk = 0; bk < 3; ++bk){
    pack_perm_kernel<16,16><<<9, 256, 0, stream>>>(P.l1w1[bk], pkL1 + (size_t)(bk*2+0)*16*144, 16);
    pack_perm_kernel<16,16><<<9, 256, 0, stream>>>(P.l1w2[bk], pkL1 + (size_t)(bk*2+1)*16*144, 16);
  }

  // ---- hyper trunk ----
  trunk_kernel<<<BATCH, 128, 0, stream>>>(P.z, P.tW1, P.tb1, P.tW2, P.tb2, h16);

  // ---- conv1 + GN + ReLU ----
  hconv_kernel<3,16,16,32,32,1,3, true,false,true,false><<<BATCH,256,0,stream>>>(
      P.x, pkC1, P.n1g, P.n1b, nullptr, bufA);

  // ---- layer1: 3 static blocks ----
  for (int bk = 0; bk < 3; ++bk){
    hconv_kernel<16,16,16,32,32,1,3, true,false,true,false><<<BATCH,256,0,stream>>>(
        bufA, pkL1 + (size_t)(bk*2+0)*16*144, P.l1g1[bk], P.l1b1[bk], nullptr, bufB);
    hconv_kernel<16,16,16,32,32,1,3, true,true,true,false><<<BATCH,256,0,stream>>>(
        bufB, pkL1 + (size_t)(bk*2+1)*16*144, P.l1g2[bk], P.l1b2[bk], bufA, bufA);
  }

  // ---- layer2 ----
  { // block 0: 16 -> 32, stride 2, hyper 1x1 shortcut
    const HyperP& h = P.l2[0];
    wgen_kernel<144,32,9><<<32*32,128,0,stream>>>(h16, h.head1, h.base1, wt);
    hconv_kernel<16,16,32,32,32,2,3, true,false,true,true><<<BATCH,256,0,stream>>>(
        bufA, wt, h.g1, h.b1, nullptr, bufB);
    wgen_kernel<16,32,1><<<32*32,128,0,stream>>>(h16, h.sch, h.scb, wt);
    hconv_kernel<16,16,32,32,32,2,1, false,false,false,true><<<BATCH,256,0,stream>>>(
        bufA, wt, nullptr, nullptr, nullptr, bufC);
    wgen_kernel<288,32,9><<<32*32,128,0,stream>>>(h16, h.head2, h.base2, wt);
    hconv_kernel<32,32,32,16,16,1,3, true,true,true,true><<<BATCH,256,0,stream>>>(
        bufB, wt, h.g2, h.b2, bufC, bufA);
  }
  for (int bk = 1; bk < 3; ++bk){ // blocks 1,2: 32 -> 32, identity shortcut
    const HyperP& h = P.l2[bk];
    wgen_kernel<288,32,9><<<32*32,128,0,stream>>>(h16, h.head1, h.base1, wt);
    hconv_kernel<32,32,32,16,16,1,3, true,false,true,true><<<BATCH,256,0,stream>>>(
        bufA, wt, h.g1, h.b1, nullptr, bufB);
    wgen_kernel<288,32,9><<<32*32,128,0,stream>>>(h16, h.head2, h.base2, wt);
    hconv_kernel<32,32,32,16,16,1,3, true,true,true,true><<<BATCH,256,0,stream>>>(
        bufB, wt, h.g2, h.b2, bufA, bufA);
  }

  // ---- layer3 ----
  { // block 0: 32 -> 64, stride 2, hyper 1x1 shortcut
    const HyperP& h = P.l3[0];
    wgen_kernel<288,64,9><<<32*64,128,0,stream>>>(h16, h.head1, h.base1, wt);
    hconv_kernel<32,32,64,16,16,2,3, true,false,true,true><<<BATCH,256,0,stream>>>(
        bufA, wt, h.g1, h.b1, nullptr, bufB);
    wgen_kernel<32,64,1><<<32*64,128,0,stream>>>(h16, h.sch, h.scb, wt);
    hconv_kernel<32,32,64,16,16,2,1, false,false,false,true><<<BATCH,256,0,stream>>>(
        bufA, wt, nullptr, nullptr, nullptr, bufC);
    wgen_kernel<576,64,9><<<32*64,128,0,stream>>>(h16, h.head2, h.base2, wt);
    hconv_kernel<64,64,64,8,8,1,3, true,true,true,true><<<BATCH,256,0,stream>>>(
        bufB, wt, h.g2, h.b2, bufC, bufA);
  }
  for (int bk = 1; bk < 3; ++bk){ // blocks 1,2: 64 -> 64, identity shortcut
    const HyperP& h = P.l3[bk];
    wgen_kernel<576,64,9><<<32*64,128,0,stream>>>(h16, h.head1, h.base1, wt);
    hconv_kernel<64,64,64,8,8,1,3, true,false,true,true><<<BATCH,256,0,stream>>>(
        bufA, wt, h.g1, h.b1, nullptr, bufB);
    wgen_kernel<576,64,9><<<32*64,128,0,stream>>>(h16, h.head2, h.base2, wt);
    hconv_kernel<64,64,64,8,8,1,3, true,true,true,true><<<BATCH,256,0,stream>>>(
        bufB, wt, h.g2, h.b2, bufA, bufA);
  }

  // ---- head: avgpool -> hyper fc ----
  avgpool_kernel<<<BATCH, 64, 0, stream>>>(bufA, feat);
  wgen_kernel<64,100,1><<<32*100,128,0,stream>>>(h16, P.fch, P.fcb, wtfc);
  fc_kernel<<<BATCH, 128, 0, stream>>>(wtfc, feat, (float*)d_out);
}
